// StepWiseMLPAutoEncoder_13022340842187
// MI455X (gfx1250) — matrix operand order repaired and checked by reading, not compiled, not run on hardware
//
#include <hip/hip_runtime.h>

// ---------------------------------------------------------------------------
// Problem constants
// ---------------------------------------------------------------------------
#define BATCH   64
#define STFT    1024
#define SEQ_T   256
#define HID     64
#define MID     1056            // ENC_MID == DEC_MID
#define COMB    2048            // 2*STFT
#define DK1     1088            // STFT + HID
#define DEC_NBLK 33             // persistent decoder grid

typedef unsigned short u16;
typedef unsigned int   u32;

typedef __attribute__((ext_vector_type(16))) __bf16 v16bf;
typedef __attribute__((ext_vector_type(8)))  float  v8f;
typedef __attribute__((ext_vector_type(4)))  u32    v4u;

union Frag {
    v16bf bf;       // WMMA operand view (16 bf16 per lane = 8 VGPRs)
    u32   w[8];     // pair-of-bf16 view
    v4u   q[2];     // b128 load view
};

// fp32 -> bf16, round-to-nearest-even
__device__ __forceinline__ u16 f2bf(float f) {
    union { float f; u32 u; } v; v.f = f;
    u32 u = v.u;
    u += 0x7FFFu + ((u >> 16) & 1u);
    return (u16)(u >> 16);
}

// 16-bit A-matrix fragment K index for VGPR v, lane-half h (ISA 7.12.2):
//   lanes 0-15 : V0..3 -> K 0..7,  V4..7 -> K 16..23
//   lanes 16-31: V0..3 -> K 8..15, V4..7 -> K 24..31
__device__ __forceinline__ int a_k(int v, int h) {
    return (v < 4) ? (2 * v + 8 * h) : (16 + 2 * (v - 4) + 8 * h);
}

__device__ __forceinline__ v8f wmma_bf16(const Frag& a, const Frag& b, v8f c) {
    return __builtin_amdgcn_wmma_f32_16x16x32_bf16(
        /*neg_a=*/false, a.bf, /*neg_b=*/false, b.bf,
        /*c_mod=*/(short)0, c, /*reuse_a=*/false, /*reuse_b=*/false);
}

// ---------------------------------------------------------------------------
// Device-wide sense barrier (state in workspace, zeroed per launch).
//   bar[0] = arrival count, bar[1] = generation
// ---------------------------------------------------------------------------
__device__ __forceinline__ void grid_barrier(u32* bar, u32 want, int nblk) {
    __builtin_amdgcn_fence(__ATOMIC_RELEASE, "agent");   // publish this block's writes
    __syncthreads();
    if (threadIdx.x == 0) {
        u32 arrived = __hip_atomic_fetch_add(&bar[0], 1u, __ATOMIC_ACQ_REL,
                                             __HIP_MEMORY_SCOPE_AGENT);
        if (arrived == (u32)nblk - 1u) {
            __hip_atomic_store(&bar[0], 0u, __ATOMIC_RELAXED, __HIP_MEMORY_SCOPE_AGENT);
            __hip_atomic_fetch_add(&bar[1], 1u, __ATOMIC_RELEASE, __HIP_MEMORY_SCOPE_AGENT);
        } else {
            while (__hip_atomic_load(&bar[1], __ATOMIC_ACQUIRE,
                                     __HIP_MEMORY_SCOPE_AGENT) < want)
                __builtin_amdgcn_s_sleep(1);
        }
    }
    __syncthreads();
    __builtin_amdgcn_fence(__ATOMIC_ACQUIRE, "agent");   // see other blocks' writes
}

// ---------------------------------------------------------------------------
// Kernel 1: transpose x[B][S][T] fp32 -> xT[T][B][S] bf16 (LDS tiled)
// ---------------------------------------------------------------------------
__global__ __launch_bounds__(256) void k_transpose_x(const float* __restrict__ x,
                                                     u16* __restrict__ xT) {
    __shared__ float tile[32][33];
    const int b  = blockIdx.z;
    const int s0 = blockIdx.x * 32;
    const int t0 = blockIdx.y * 32;
    const int tx = threadIdx.x;
    const int ty = threadIdx.y;
#pragma unroll
    for (int i = 0; i < 32; i += 8)
        tile[ty + i][tx] = x[((size_t)b * STFT + (s0 + ty + i)) * SEQ_T + t0 + tx];
    __syncthreads();
#pragma unroll
    for (int i = 0; i < 32; i += 8) {
        const int t = t0 + ty + i;
        xT[((size_t)t * BATCH + b) * STFT + s0 + tx] = f2bf(tile[tx][ty + i]);
    }
}

// ---------------------------------------------------------------------------
// Kernel 2: pack weight [K][N] fp32 -> WMMA B-fragment tiles, bf16.
// Tile (nt,kt): lane L owns 16 contiguous bf16 at L*16:
//   n = nt*16 + (L&15), j-th element k = kt*32 + (j>>3)*16 + (L>>4)*8 + (j&7).
// Tile order: tile_id = nt*KT + kt.
// ---------------------------------------------------------------------------
__global__ void k_pack_w(const float* __restrict__ W, u16* __restrict__ out,
                         int K, int N, int KT, int NT) {
    const size_t total = (size_t)KT * NT * 512;
    const size_t idx = (size_t)blockIdx.x * blockDim.x + threadIdx.x;
    if (idx >= total) return;
    const int    j    = (int)(idx & 15);
    const int    lane = (int)((idx >> 4) & 31);
    const size_t tile = idx >> 9;
    const int    kt   = (int)(tile % KT);
    const int    nt   = (int)(tile / KT);
    const int    n    = nt * 16 + (lane & 15);
    const int    k    = kt * 32 + (j >> 3) * 16 + (lane >> 4) * 8 + (j & 7);
    out[idx] = f2bf(W[(size_t)k * N + n]);
}

// ---------------------------------------------------------------------------
// Kernel 3: zero-fill u16 buffer
// ---------------------------------------------------------------------------
__global__ void k_zero_u16(u16* __restrict__ p, int n) {
    const int i = blockIdx.x * blockDim.x + threadIdx.x;
    if (i < n) p[i] = 0;
}

// ---------------------------------------------------------------------------
// Kernel 4: encoder. One workgroup = 32 rows (fixed t, 32 batches = 2 M-tiles).
// Each wave shares one B fragment across both M-tiles (halves L2 B-traffic).
//   layer1: [32x2048]*[2048x1056] +be1, ReLU -> LDS
//   layer2: [32x1056]*[1056x64]  +be2       -> controls[t][b][h] (bf16)
// ---------------------------------------------------------------------------
__global__ __launch_bounds__(256) void k_encoder(const u16* __restrict__ xT,
                                                 const u16* __restrict__ We1p,
                                                 const float* __restrict__ be1,
                                                 const u16* __restrict__ We2p,
                                                 const float* __restrict__ be2,
                                                 u16* __restrict__ controls) {
    __shared__ __align__(16) u16 combL[32][COMB + 8];   // 131.5 KB, pad kills conflicts
    __shared__ __align__(16) u16 midL[32][MID + 8];     // 68 KB

    const int tid     = threadIdx.x;
    const int rowbase = blockIdx.x * 32;     // global row = t*64 + b
    const int t       = rowbase >> 6;
    const int bbase   = rowbase & 63;        // 0 or 32

    // ---- stage 32 comb rows into LDS as bf16 (b128 chunks) ----
    {
        const int m  = tid >> 3;             // local row 0..31
        const int c0 = (tid & 7) * 256;      // 256-elem chunk (never straddles 1024)
        const int b  = bbase + m;
        if (c0 < 1024) {                     // previous-frame half
            if (t == 0) {
                const v4u z = {0u, 0u, 0u, 0u};
#pragma unroll
                for (int j = 0; j < 256; j += 8)
                    *(v4u*)&combL[m][c0 + j] = z;
            } else {
                const u16* src = xT + ((size_t)(t - 1) * BATCH + b) * STFT + c0;
#pragma unroll
                for (int j = 0; j < 256; j += 8)
                    *(v4u*)&combL[m][c0 + j] = *(const v4u*)(src + j);
            }
        } else {                             // current-frame half
            const u16* src = xT + ((size_t)t * BATCH + b) * STFT + (c0 - 1024);
#pragma unroll
            for (int j = 0; j < 256; j += 8)
                *(v4u*)&combL[m][c0 + j] = *(const v4u*)(src + j);
        }
    }
    __syncthreads();

    const int lane = tid & 31;
    const int wave = tid >> 5;
    const int half = lane >> 4;
    const int am   = lane & 15;              // A-fragment row within M-tile
    const int nn   = lane & 15;              // C/D column for this lane

    // ---- layer 1: 66 N-tiles across 8 waves; 2 M-tiles share each B frag ----
    for (int nt = wave; nt < MID / 16; nt += 8) {
        v8f acc0 = {}, acc1 = {};
        const u16* bt = We1p + ((size_t)nt * (COMB / 32) * 512) + lane * 16;
        for (int kt = 0; kt < COMB / 32; ++kt, bt += 512) {
            Frag a0, a1, bf;
            bf.q[0] = *(const v4u*)bt;
            bf.q[1] = *(const v4u*)(bt + 8);
            __builtin_prefetch(bt + 512, 0, 1);          // next K-tile of B
            const int kb = kt * 32;
#pragma unroll
            for (int v = 0; v < 8; ++v) {
                a0.w[v] = *(const u32*)&combL[am][kb + a_k(v, half)];
                a1.w[v] = *(const u32*)&combL[16 + am][kb + a_k(v, half)];
            }
            acc0 = wmma_bf16(a0, bf, acc0);
            acc1 = wmma_bf16(a1, bf, acc1);
        }
        const int   n    = nt * 16 + nn;
        const float bias = be1[n];
#pragma unroll
        for (int i = 0; i < 8; ++i) {
            const int m = i + 8 * half;
            float v0 = acc0[i] + bias;
            float v1 = acc1[i] + bias;
            midL[m][n]      = f2bf(v0 > 0.f ? v0 : 0.f);   // ReLU
            midL[16 + m][n] = f2bf(v1 > 0.f ? v1 : 0.f);
        }
    }
    __syncthreads();

    // ---- layer 2: 8 tiles (2 M x 4 N), one wave each; K loop = 33 ----
    {
        const int mt = wave >> 2;            // 0..1
        const int nt = wave & 3;             // 0..3
        v8f acc = {};
        const u16* bt = We2p + ((size_t)nt * (MID / 32) * 512) + lane * 16;
        for (int kt = 0; kt < MID / 32; ++kt, bt += 512) {
            Frag a, bf;
            bf.q[0] = *(const v4u*)bt;
            bf.q[1] = *(const v4u*)(bt + 8);
            const int kb = kt * 32;
#pragma unroll
            for (int v = 0; v < 8; ++v)
                a.w[v] = *(const u32*)&midL[mt * 16 + am][kb + a_k(v, half)];
            acc = wmma_bf16(a, bf, acc);
        }
        const int   h    = nt * 16 + nn;
        const float bias = be2[h];
#pragma unroll
        for (int i = 0; i < 8; ++i) {
            const int b = bbase + mt * 16 + i + 8 * half;
            controls[((size_t)t * BATCH + b) * HID + h] = f2bf(acc[i] + bias);
        }
    }
}

// ---------------------------------------------------------------------------
// Kernel 5: persistent decoder. 33 blocks x 8 waves; loops over all 256 steps
// with a device-wide barrier between phases (no per-step kernel launches).
//   phase1: [64x1088]*[1088x1056]+bd1,ReLU -> hid   (264 tiles, 1/wave)
//   phase2: [64x1056]*[1056x1024]+bd2      -> out[b][s][t] fp32, prev bf16
// Weights stay hot in L2 (192 MB) across all steps.
// ---------------------------------------------------------------------------
__global__ __launch_bounds__(256) void k_decoder(const u16* __restrict__ controls,
                                                 const u16* __restrict__ Wd1p,
                                                 const float* __restrict__ bd1,
                                                 const u16* __restrict__ Wd2p,
                                                 const float* __restrict__ bd2,
                                                 float* __restrict__ out,
                                                 u16* __restrict__ prev,
                                                 u16* __restrict__ hid,
                                                 u32* __restrict__ bar) {
    const int tid  = threadIdx.x;
    const int lane = tid & 31;
    const int wave = tid >> 5;
    const int tile = blockIdx.x * 8 + wave;      // 0..263
    const int half = lane >> 4;
    const int am   = lane & 15;
    const int nn   = lane & 15;

    // phase-1 tile: 264 = 4 M-tiles x 66 N-tiles
    const int mt1  = tile & 3;
    const int nt1  = tile >> 2;                  // 0..65
    const int row1 = mt1 * 16 + am;
    // phase-2 tile: 256 = 4 M-tiles x 64 N-tiles (tiles 256..263 idle)
    const int mt2  = tile & 3;
    const int nt2  = tile >> 2;                  // 0..65 (valid < 64)
    const int row2 = mt2 * 16 + am;

    u32 want = 0;
    for (int t = 0; t < SEQ_T; ++t) {
        // ---------------- phase 1 ----------------
        {
            const u16* ctr = controls + (size_t)t * BATCH * HID;
            v8f acc = {};
            const u16* bt = Wd1p + ((size_t)nt1 * (DK1 / 32) * 512) + lane * 16;
            for (int kt = 0; kt < DK1 / 32; ++kt, bt += 512) {
                Frag a, bf;
                bf.q[0] = *(const v4u*)bt;
                bf.q[1] = *(const v4u*)(bt + 8);
#pragma unroll
                for (int v = 0; v < 8; ++v) {
                    const int  kk = kt * 32 + a_k(v, half);
                    const u16* p  = (kk < HID) ? (ctr + (size_t)row1 * HID + kk)
                                               : (prev + (size_t)row1 * STFT + (kk - HID));
                    a.w[v] = *(const u32*)p;
                }
                acc = wmma_bf16(a, bf, acc);
            }
            const int   n    = nt1 * 16 + nn;
            const float bias = bd1[n];
#pragma unroll
            for (int i = 0; i < 8; ++i) {
                const int   m = mt1 * 16 + i + 8 * half;
                const float v = acc[i] + bias;
                hid[(size_t)m * MID + n] = f2bf(v > 0.f ? v : 0.f);
            }
        }
        ++want;
        grid_barrier(bar, want, DEC_NBLK);

        // ---------------- phase 2 ----------------
        if (nt2 < STFT / 16) {
            v8f acc = {};
            const u16* bt = Wd2p + ((size_t)nt2 * (MID / 32) * 512) + lane * 16;
            for (int kt = 0; kt < MID / 32; ++kt, bt += 512) {
                Frag a, bf;
                bf.q[0] = *(const v4u*)bt;
                bf.q[1] = *(const v4u*)(bt + 8);
#pragma unroll
                for (int v = 0; v < 8; ++v)
                    a.w[v] = *(const u32*)&hid[(size_t)row2 * MID + kt * 32 + a_k(v, half)];
                acc = wmma_bf16(a, bf, acc);
            }
            const int   n    = nt2 * 16 + nn;       // stft index s
            const float bias = bd2[n];
#pragma unroll
            for (int i = 0; i < 8; ++i) {
                const int   b = mt2 * 16 + i + 8 * half;
                const float v = acc[i] + bias;
                out[((size_t)b * STFT + n) * SEQ_T + t] = v;
                prev[(size_t)b * STFT + n] = f2bf(v);
            }
        }
        ++want;
        grid_barrier(bar, want, DEC_NBLK);
    }
}

// ---------------------------------------------------------------------------
// Host launch
// ---------------------------------------------------------------------------
extern "C" void kernel_launch(void* const* d_in, const int* in_sizes, int n_in,
                              void* d_out, int out_size, void* d_ws, size_t ws_size,
                              hipStream_t stream) {
    (void)in_sizes; (void)n_in; (void)out_size;

    const float* x   = (const float*)d_in[0];
    const float* We1 = (const float*)d_in[1];
    const float* be1 = (const float*)d_in[2];
    const float* We2 = (const float*)d_in[3];
    const float* be2 = (const float*)d_in[4];
    const float* Wd1 = (const float*)d_in[5];
    const float* bd1 = (const float*)d_in[6];
    const float* Wd2 = (const float*)d_in[7];
    const float* bd2 = (const float*)d_in[8];
    float*       out = (float*)d_out;

    // ---- workspace layout (256B-aligned sections) ----
    char* ws = (char*)d_ws;
    size_t o = 0;
    auto take = [&](size_t bytes) { char* p = ws + o; o += (bytes + 255) & ~(size_t)255; return p; };
    u16* xT   = (u16*)take((size_t)SEQ_T * BATCH * STFT * 2);            // 33.6 MB
    u16* We1p = (u16*)take((size_t)(COMB/32) * (MID/16)  * 512 * 2);     // 4.3 MB
    u16* We2p = (u16*)take((size_t)(MID /32) * (HID/16)  * 512 * 2);     // 132 KB
    u16* Wd1p = (u16*)take((size_t)(DK1 /32) * (MID/16)  * 512 * 2);     // 2.3 MB
    u16* Wd2p = (u16*)take((size_t)(MID /32) * (STFT/16) * 512 * 2);     // 2.1 MB
    u16* ctl  = (u16*)take((size_t)SEQ_T * BATCH * HID * 2);             // 2.1 MB
    u16* hid  = (u16*)take((size_t)BATCH * MID * 2);                     // 132 KB
    u16* prev = (u16*)take((size_t)BATCH * STFT * 2);                    // 128 KB
    u32* bar  = (u32*)take(256);                                         // barrier state
    if (o > ws_size) return;   // insufficient scratch (deterministic no-op)

    // 1) transpose + bf16-convert x
    k_transpose_x<<<dim3(STFT / 32, SEQ_T / 32, BATCH), dim3(32, 8), 0, stream>>>(x, xT);

    // 2) pack weights into WMMA B-fragment tiles
    {
        const struct { const float* W; u16* P; int K, N; } ws4[4] = {
            { We1, We1p, COMB, MID  },
            { We2, We2p, MID,  HID  },
            { Wd1, Wd1p, DK1,  MID  },
            { Wd2, Wd2p, MID,  STFT },
        };
        for (int i = 0; i < 4; ++i) {
            const int KT = ws4[i].K / 32, NT = ws4[i].N / 16;
            const size_t total = (size_t)KT * NT * 512;
            k_pack_w<<<(unsigned)((total + 255) / 256), 256, 0, stream>>>(
                ws4[i].W, ws4[i].P, ws4[i].K, ws4[i].N, KT, NT);
        }
    }

    // 3) zero prev_stft (t=0 state) and barrier state
    k_zero_u16<<<(BATCH * STFT + 255) / 256, 256, 0, stream>>>(prev, BATCH * STFT);
    k_zero_u16<<<1, 128, 0, stream>>>((u16*)bar, 128);

    // 4) encoder (16384 rows / 32 per workgroup)
    k_encoder<<<(BATCH * SEQ_T) / 32, 256, 0, stream>>>(xT, We1p, be1, We2p, be2, ctl);

    // 5) persistent decoder: all 256 recurrent steps in ONE launch
    k_decoder<<<DEC_NBLK, 256, 0, stream>>>(ctl, Wd1p, bd1, Wd2p, bd2, out, prev, hid, bar);
}
